// NuFFTNarrow_36996848288300
// MI455X (gfx1250) — matrix operand order repaired and checked by reading, not compiled, not run on hardware
//
#include <hip/hip_runtime.h>
#include <math.h>

// ---------------------------------------------------------------------------
// NUFFT forward for MI455X (gfx1250, wave32).
//   prep    : fftshift + KB grid-correction + zero-pad + roll  -> P (complex)
//   fft1k   : 1024-pt complex FFT lines (rows then cols), radix 16*16*4,
//             radix-16 stages done as V_WMMA_F32_16X16X32_F16 complex GEMMs
//   degrid  : 6x6 Kaiser-Bessel gather per visibility (L2-resident F grid)
// ---------------------------------------------------------------------------

typedef __attribute__((ext_vector_type(16))) _Float16 v16h;
typedef __attribute__((ext_vector_type(8)))  float    v8f;

#define NCHAN 4
#define NPIX  512
#define GRD   1024
#define NVIS  200000
#define JW    6
#define ALPHA_KB 14.04f            // 2.34 * J
#define PI_F 3.14159265358979f

// Abramowitz & Stegun 9.8.1 / 9.8.2 modified Bessel I0
__device__ __forceinline__ float bessel_i0f(float x) {
    float ax = fabsf(x);
    if (ax < 3.75f) {
        float t = ax / 3.75f; t *= t;
        return 1.0f + t*(3.5156229f + t*(3.0899424f + t*(1.2067492f +
                     t*(0.2659732f + t*(0.0360768f + t*0.0045813f)))));
    }
    float t = 3.75f / ax;
    return (expf(ax) * __frsqrt_rn(ax)) *
           (0.39894228f + t*(0.01328592f + t*(0.00225319f + t*(-0.00157565f +
            t*(0.00916281f + t*(-0.02057706f + t*(0.02635537f +
            t*(-0.01647633f + t*0.00392377f))))))));
}

// 1 / kb_ft(f):   kb_ft = J*sinh(z)/(z*I0(alpha)),  z = sqrt(a^2-(pi J f)^2)
__device__ __forceinline__ float kbft_inv(float f, float i0a) {
    float pjf = PI_F * (float)JW * f;
    float z   = sqrtf(fmaxf(ALPHA_KB*ALPHA_KB - pjf*pjf, 1e-12f));
    return (z * i0a) / ((float)JW * sinhf(z));
}

__device__ __forceinline__ float kb_kernel_dev(float u, float inv_i0a) {
    float r = 2.0f * u / (float)JW;
    float x = 1.0f - r * r;
    if (x <= 0.0f) return 0.0f;
    return bessel_i0f(ALPHA_KB * sqrtf(x)) * inv_i0a;
}

// ---------------------------------------------------------------------------
// Kernel 1: build padded/rolled/corrected complex grid P[c][1024][1024].
// Combined index math: nonzero only for g in [0,256) u [768,1024) per axis;
//   cube row = g (g<256) or g-512 (g>=768); correction arg f = g/G or (g-G)/G.
// ---------------------------------------------------------------------------
__global__ void nufft_prep(const float* __restrict__ cube,
                           float2* __restrict__ P) {
    int idx = blockIdx.x * 256 + threadIdx.x;
    if (idx >= NCHAN * GRD * GRD) return;
    int gx = idx & (GRD - 1);
    int gy = (idx >> 10) & (GRD - 1);
    int c  = idx >> 20;

    float2 v = make_float2(0.0f, 0.0f);
    bool iny = (gy < NPIX/2) | (gy >= GRD - NPIX/2);
    bool inx = (gx < NPIX/2) | (gx >= GRD - NPIX/2);
    if (iny & inx) {
        int   cy = (gy < NPIX/2) ? gy : gy - (GRD - NPIX);
        int   cx = (gx < NPIX/2) ? gx : gx - (GRD - NPIX);
        float fy = ((gy < NPIX/2) ? (float)gy : (float)(gy - GRD)) * (1.0f/(float)GRD);
        float fx = ((gx < NPIX/2) ? (float)gx : (float)(gx - GRD)) * (1.0f/(float)GRD);
        float i0a = bessel_i0f(ALPHA_KB);
        v.x = cube[((size_t)c * NPIX + cy) * NPIX + cx] *
              kbft_inv(fy, i0a) * kbft_inv(fx, i0a);
    }
    P[idx] = v;
}

// ---------------------------------------------------------------------------
// Kernel 2: 1024-point complex FFT, one wave per line, 4 waves / block.
// N = 16*64, 64 = 16*4:
//   S1: G[k][c]   = sum_r x[64r+c] W16^{rk}           (WMMA GEMM, K=32 packed)
//       twiddle W1024^{ck}
//   S2: Z[k'][k,c'] = sum_{r'} G[k][4r'+c'] W16^{r'k'} (WMMA GEMM)
//       twiddle W64^{c'k'}
//   S3: radix-4 over c', output index m = 256j' + 16k' + k.
// A operand (16x32 f16) = [cos | sin] (Re path) and [-sin | cos] (Im path);
// B operand (32x16 f16) = [Xre ; Xim]; accumulate f32.
// ---------------------------------------------------------------------------
__global__ __launch_bounds__(128) void nufft_fft1k(const float2* __restrict__ in,
                                                   float2* __restrict__ out,
                                                   int col_pass) {
    __shared__ float smem[4 * 4 * 1024];   // [wave][4 planes][1024] = 64 KB
    const int lane = threadIdx.x & 31;
    const int wave = threadIdx.x >> 5;
    const int half = lane >> 4;            // this code: lanes 16-31 hold upper K half (K16..31)
    const int nl   = lane & 15;            // B/D column index N = lane % 16

    float* bufr0 = &smem[wave * 4096 + 0];
    float* bufi0 = &smem[wave * 4096 + 1024];
    float* bufr1 = &smem[wave * 4096 + 2048];
    float* bufi1 = &smem[wave * 4096 + 3072];

    const int line = blockIdx.x * 4 + wave;        // 0..4095 = chan*1024 + line
    const int c    = line >> 10;
    const int r    = line & 1023;
    size_t base; int stride;
    if (col_pass) { base = (size_t)c * GRD * GRD + r;        stride = GRD; }
    else          { base = ((size_t)c * GRD + r) * GRD;      stride = 1;   }

    // global -> LDS (coalesced row pass; 8KB-strided col pass, L2 resident)
    #pragma unroll
    for (int i = 0; i < 32; ++i) {
        int e = lane + 32 * i;
        float2 v = in[base + (size_t)e * stride];
        bufr0[e] = v.x; bufi0[e] = v.y;
    }
    __syncthreads();

    // Constant DFT16 A-fragments (16x32 f16, contiguous K layout:
    // lanes 0-15 slots 0..15 -> K0..15; lanes 16-31 slots 0..15 -> K16..31)
    v16h ar, ai;
    #pragma unroll
    for (int s = 0; s < 16; ++s) {
        int kk = s + (half ? 16 : 0);
        int kr = (kk < 16) ? kk : kk - 16;
        float th = (2.0f * PI_F / 16.0f) * (float)((nl * kr) & 15);
        float cs = cosf(th), sn = sinf(th);
        ar[s] = (_Float16)((kk < 16) ?  cs : sn);   // Yr = cos*xr + sin*xi
        ai[s] = (_Float16)((kk < 16) ? -sn : cs);   // Yi = -sin*xr + cos*xi
    }
    const v8f zacc = {};

    // ---- Stage 1: B = D16 * A,  A[r][c] = x[64r + c], then W1024^{ck} ------
    #pragma unroll
    for (int t = 0; t < 4; ++t) {
        int cc = 16 * t + nl;                       // this lane's column
        const float* src = half ? bufi0 : bufr0;    // B layout: K = s + 16*half
        v16h b;
        #pragma unroll
        for (int s = 0; s < 16; ++s) b[s] = (_Float16)src[64 * s + cc];
        v8f dr = __builtin_amdgcn_wmma_f32_16x16x32_f16(false, ar, false, b,
                                                        (short)0, zacc, false, false);
        v8f di = __builtin_amdgcn_wmma_f32_16x16x32_f16(false, ai, false, b,
                                                        (short)0, zacc, false, false);
        #pragma unroll
        for (int g = 0; g < 8; ++g) {
            int k = g + 8 * half;                   // D layout: M = g + 8*half
            float ph = (2.0f * PI_F / 1024.0f) * (float)((cc * k) & 1023);
            float sn, cs; __sincosf(ph, &sn, &cs);
            bufr1[64 * k + cc] = dr[g] * cs + di[g] * sn;   // * e^{-i ph}
            bufi1[64 * k + cc] = di[g] * cs - dr[g] * sn;
        }
    }
    __syncthreads();

    // ---- Stage 2: inner DFT16 over r' of G[k][4r'+c'], then W64^{c'k'} -----
    #pragma unroll
    for (int t = 0; t < 4; ++t) {
        int q  = 16 * t + nl;                       // column = (k, c') pair
        int kq = q >> 2, cp = q & 3;
        const float* src = half ? bufi1 : bufr1;
        v16h b;
        #pragma unroll
        for (int s = 0; s < 16; ++s) b[s] = (_Float16)src[64 * kq + 4 * s + cp];
        v8f dr = __builtin_amdgcn_wmma_f32_16x16x32_f16(false, ar, false, b,
                                                        (short)0, zacc, false, false);
        v8f di = __builtin_amdgcn_wmma_f32_16x16x32_f16(false, ai, false, b,
                                                        (short)0, zacc, false, false);
        #pragma unroll
        for (int g = 0; g < 8; ++g) {
            int kp = g + 8 * half;
            float ph = (2.0f * PI_F / 64.0f) * (float)((cp * kp) & 63);
            float sn, cs; __sincosf(ph, &sn, &cs);
            bufr0[kp * 64 + kq * 4 + cp] = dr[g] * cs + di[g] * sn;
            bufi0[kp * 64 + kq * 4 + cp] = di[g] * cs - dr[g] * sn;
        }
    }
    __syncthreads();

    // ---- Stage 3: radix-4 over c', write X[256j' + 16k' + k] ---------------
    #pragma unroll
    for (int i = 0; i < 8; ++i) {
        int p  = lane + 32 * i;                     // 256 (k',k) pairs per line
        int kp = p >> 4, k = p & 15;
        float zr[4], zi[4];
        #pragma unroll
        for (int cq = 0; cq < 4; ++cq) {
            zr[cq] = bufr0[kp * 64 + k * 4 + cq];
            zi[cq] = bufi0[kp * 64 + k * 4 + cq];
        }
        float xr[4], xi[4];
        xr[0] = zr[0] + zr[1] + zr[2] + zr[3];  xi[0] = zi[0] + zi[1] + zi[2] + zi[3];
        xr[1] = zr[0] + zi[1] - zr[2] - zi[3];  xi[1] = zi[0] - zr[1] - zi[2] + zr[3];
        xr[2] = zr[0] - zr[1] + zr[2] - zr[3];  xi[2] = zi[0] - zi[1] + zi[2] - zi[3];
        xr[3] = zr[0] - zi[1] - zr[2] + zi[3];  xi[3] = zi[0] + zr[1] - zi[2] - zr[3];
        #pragma unroll
        for (int jp = 0; jp < 4; ++jp) {
            int m = 256 * jp + 16 * kp + k;
            out[base + (size_t)m * stride] = make_float2(xr[jp], xi[jp]);
        }
    }
}

// ---------------------------------------------------------------------------
// Kernel 3: degridding. One thread per visibility; 4x36 complex taps gathered
// from the L2-resident F grid; separable KB weights; scale by CELL_SIZE^2.
// ---------------------------------------------------------------------------
__global__ void nufft_degrid(const float* __restrict__ uu,
                             const float* __restrict__ vv,
                             const float2* __restrict__ F,
                             float2* __restrict__ outv) {
    int m = blockIdx.x * 256 + threadIdx.x;
    if (m >= NVIS) return;

    // tu = uu * 1000 * dl * G, dl = 0.005 arcsec
    const float KT = 5120.0f * PI_F / 648000.0f;
    float tu = uu[m] * KT;
    float tv = vv[m] * KT;
    float fu = floorf(tu), fv = floorf(tv);
    float inv_i0a = 1.0f / bessel_i0f(ALPHA_KB);

    int   iu[JW], iv[JW];
    float wu[JW], wv[JW];
    #pragma unroll
    for (int i = 0; i < JW; ++i) {
        float off = (float)(i - 2);                 // offs = arange(6) - 2
        wu[i] = kb_kernel_dev(tu - (fu + off), inv_i0a);
        wv[i] = kb_kernel_dev(tv - (fv + off), inv_i0a);
        iu[i] = ((int)fu + (i - 2)) & (GRD - 1);
        iv[i] = ((int)fv + (i - 2)) & (GRD - 1);
    }

    #pragma unroll
    for (int c = 0; c < NCHAN; ++c) {
        float sr = 0.0f, si = 0.0f;
        #pragma unroll
        for (int i = 0; i < JW; ++i) {
            const float2* row = F + ((size_t)(c * GRD + iv[i])) * GRD;
            float wvi = wv[i];
            #pragma unroll
            for (int j = 0; j < JW; ++j) {
                float2 V = row[iu[j]];
                float  w = wvi * wu[j];
                sr = fmaf(V.x, w, sr);
                si = fmaf(V.y, w, si);
            }
        }
        outv[(size_t)c * NVIS + m] = make_float2(sr * 2.5e-5f, si * 2.5e-5f);
    }
}

// ---------------------------------------------------------------------------
extern "C" void kernel_launch(void* const* d_in, const int* in_sizes, int n_in,
                              void* d_out, int out_size, void* d_ws, size_t ws_size,
                              hipStream_t stream) {
    (void)in_sizes; (void)n_in; (void)out_size; (void)ws_size;
    const float* cube = (const float*)d_in[0];
    const float* uu   = (const float*)d_in[1];
    const float* vv   = (const float*)d_in[2];

    float2* bufA = (float2*)d_ws;                        // pad grid, then final F
    float2* bufB = bufA + (size_t)NCHAN * GRD * GRD;     // row-FFT intermediate

    int nP = NCHAN * GRD * GRD;
    nufft_prep<<<(nP + 255) / 256, 256, 0, stream>>>(cube, bufA);
    nufft_fft1k<<<1024, 128, 0, stream>>>(bufA, bufB, 0);   // rows
    nufft_fft1k<<<1024, 128, 0, stream>>>(bufB, bufA, 1);   // cols
    nufft_degrid<<<(NVIS + 255) / 256, 256, 0, stream>>>(uu, vv, bufA,
                                                         (float2*)d_out);
}
